// ODE_RNN_7198365188497
// MI455X (gfx1250) — compile-verified, matrix operand order checked by reading, not run on hardware
//
#include <hip/hip_runtime.h>
#include <cmath>

// ---------------------------------------------------------------------------
// ODE-RNN (latent ODE + GRU) for gfx1250.  B=512, T=256, LATENT=64, UNITS=128.
// Persistent workgroups (32 batch rows each) scan 255 steps; all dense layers
// run on v_wmma_f32_16x16x32_f16 with f32 accumulation.  Weights (~320KB f16)
// are pre-swizzled into WMMA B-fragment order in d_ws and stay L2-resident.
// Activations use native gfx1250 transcendentals (v_tanh_f32 / v_exp_f32 /
// v_log_f32 / v_rcp_f32) -- branchless, co-executable with the WMMA pipe.
// ---------------------------------------------------------------------------

typedef __attribute__((ext_vector_type(16))) _Float16 v16h;
typedef __attribute__((ext_vector_type(8)))  float    v8f;

constexpr int B_   = 512;
constexpr int T_   = 256;
constexpr int TSTEPS = 255;       // scan length (T-1)
constexpr int LAT  = 64;
constexpr int IN_  = 32;
constexpr int MT   = 32;          // batch rows per workgroup
constexpr int NBLK = B_ / MT;     // 16 workgroups
constexpr int AS   = 160;         // LDS activation row stride (halves)
constexpr long OUTOFF = (long)B_ * TSTEPS * IN_;  // stds offset in d_out

// f16 weight-fragment base offsets (halves) inside d_ws, in use order:
// ode W0..W3, upd W0,W1, rst W0,W1, new W0,W1, dec W0,W1
constexpr int WO0 = 0;        // 64x128
constexpr int WO1 = 8192;     // 128x128
constexpr int WO2 = 24576;    // 128x128
constexpr int WO3 = 40960;    // 128x64
constexpr int WO4 = 49152;    // 160x128
constexpr int WO5 = 69632;    // 128x64
constexpr int WO6 = 77824;    // 160x128
constexpr int WO7 = 98304;    // 128x64
constexpr int WO8 = 106496;   // 160x128
constexpr int WO9 = 126976;   // 128x128
constexpr int WO10 = 143360;  // 64x128
constexpr int WO11 = 151552;  // 128x64

union ABf { uint4 q[2]; v16h v; };

struct Biases { const float* p[12]; };

// ---- fast branchless activations on native gfx1250 trans ops --------------
__device__ __forceinline__ float fexp2(float x) { return __builtin_amdgcn_exp2f(x); }
__device__ __forceinline__ float flog2(float x) { return __builtin_amdgcn_logf(x); }
__device__ __forceinline__ float frcp (float x) { return __builtin_amdgcn_rcpf(x); }

__device__ __forceinline__ float ftanh(float x) {
#if defined(__gfx1250__) && __has_builtin(__builtin_amdgcn_tanhf)
    return __builtin_amdgcn_tanhf(x);           // v_tanh_f32
#else
    // tanh(x) = 1 - 2/(exp2(2x*log2e)+1); saturates correctly at +-inf
    return 1.f - 2.f * frcp(1.f + fexp2(2.8853900817779268f * x));
#endif
}
__device__ __forceinline__ float fsigmoid(float x) {
    return frcp(1.f + fexp2(-1.4426950408889634f * x));
}
__device__ __forceinline__ float fsoftplus(float x) {
    float e  = fexp2(1.4426950408889634f * x);
    float sp = 0.6931471805599453f * flog2(1.f + e);
    return (x > 20.f) ? x : sp;                 // branchless select
}

// ---------------------------------------------------------------------------
// Prep: convert f32 KxN weight (row-major, x@W convention) into f16 fragments.
// Fragment (kt,nt) = 32 lanes x 16 halves, matching the 16-bit WMMA A/B VGPR
// layout: lane l holds col n = nt*16 + (l&15); half e maps to
// k = kt*32 + e + 8*(l>>4) + (e>=8 ? 8 : 0).
// ---------------------------------------------------------------------------
__global__ void prep_weights(const float* __restrict__ W, _Float16* __restrict__ dst,
                             int K, int N) {
    int idx = blockIdx.x * 256 + threadIdx.x;
    if (idx >= K * N) return;
    int f  = idx >> 9;          // fragment index
    int r  = idx & 511;
    int l  = r >> 4;            // lane
    int e  = r & 15;            // half within lane
    int KT = K >> 5;
    int kt = f % KT;
    int nt = f / KT;
    int n  = nt * 16 + (l & 15);
    int hi = l >> 4;
    int k  = kt * 32 + e + 8 * hi + (e >= 8 ? 8 : 0);
    dst[idx] = (_Float16)W[k * N + n];
}

// ---------------------------------------------------------------------------
// Dense layer core: act (LDS f16, stride AS) x W-fragments (global f16) with
// f32 WMMA accumulation.  Each wave computes tiles nt = colgroup, colgroup+4.
// Epilogue lambda receives the v8f accumulator and nt.
// ---------------------------------------------------------------------------
template<int K, int N, class Epi>
__device__ __forceinline__
void dense(const _Float16* __restrict__ act, const _Float16* __restrict__ w,
           int lane, int rowtile, int colgroup, Epi&& epi) {
    constexpr int KT = K / 32;
    constexpr int NT = N / 16;
    const int m  = rowtile * 16 + (lane & 15);
    const int hi = lane >> 4;
    const _Float16* arow = act + m * AS + hi * 8;
    for (int nt = colgroup; nt < NT; nt += 4) {
        v8f c = {};
#pragma unroll
        for (int kt = 0; kt < KT; ++kt) {
            ABf a, b;
            const _Float16* ap = arow + kt * 32;
            a.q[0] = *(const uint4*)(ap);
            a.q[1] = *(const uint4*)(ap + 16);
            const _Float16* wp = w + ((nt * KT + kt) * 32 + lane) * 16;
            b.q[0] = *(const uint4*)(wp);
            b.q[1] = *(const uint4*)(wp + 8);
            c = __builtin_amdgcn_wmma_f32_16x16x32_f16(false, a.v, false, b.v,
                                                       (short)0, c, false, false);
        }
        epi(c, nt);
    }
}

__global__ __launch_bounds__(256)
void ode_rnn_main(const float* __restrict__ data, const float* __restrict__ mask,
                  const float* __restrict__ mask_first, const float* __restrict__ ts,
                  Biases bz, const _Float16* __restrict__ W, float* __restrict__ out)
{
    __shared__ __align__(16) _Float16 actA[MT * AS];
    __shared__ __align__(16) _Float16 actB[MT * AS];
    __shared__ __align__(16) float    h   [MT * LAT];   // carried h / hidden_ode
    __shared__ __align__(16) float    hstd[MT * LAT];
    __shared__ __align__(16) float    obs [MT * IN_];
    __shared__ __align__(16) _Float16 u16 [MT * LAT];
    __shared__ __align__(16) _Float16 r16 [MT * LAT];
    __shared__ __align__(16) _Float16 ns16[MT * LAT];
    __shared__ __align__(16) _Float16 nd16[MT * LAT];
    __shared__ float mrow[MT];
    __shared__ float mfro[MT];

    const int tid      = threadIdx.x;
    const int lane     = tid & 31;
    const int wave     = tid >> 5;
    const int rowtile  = wave >> 2;
    const int colgroup = wave & 3;
    const int b0       = blockIdx.x * MT;
    const int hi       = lane >> 4;
    const int nl       = lane & 15;

    // ---- init carried state -------------------------------------------------
    for (int i = tid; i < MT * LAT; i += 256) { h[i] = 0.f; hstd[i] = 0.f; }
    for (int i = tid; i < MT * IN_; i += 256) {
        int r = i >> 5, c = i & 31;
        obs[i] = data[(size_t)(b0 + r) * (T_ * IN_) + c];   // data[:,0,:]
    }
    __syncthreads();

    for (int t = 0; t < TSTEPS; ++t) {
        const float dt = ts[t + 1] - ts[t];

        // ---- (1) actA <- h (f16) -------------------------------------------
        for (int i = tid; i < MT * LAT; i += 256)
            actA[(i >> 6) * AS + (i & 63)] = (_Float16)h[i];
        __syncthreads();

        // ---- ODE MLP: 64->128->128->128->64, tanh between layers -----------
        dense<64,128>(actA, W + WO0, lane, rowtile, colgroup, [&](v8f c, int nt){
            int n = nt * 16 + nl; float bv = bz.p[0][n];
#pragma unroll
            for (int v = 0; v < 8; ++v) {
                int row = rowtile * 16 + hi * 8 + v;
                actB[row * AS + n] = (_Float16)ftanh(c[v] + bv);
            }
        });
        __syncthreads();
        dense<128,128>(actB, W + WO1, lane, rowtile, colgroup, [&](v8f c, int nt){
            int n = nt * 16 + nl; float bv = bz.p[1][n];
#pragma unroll
            for (int v = 0; v < 8; ++v) {
                int row = rowtile * 16 + hi * 8 + v;
                actA[row * AS + n] = (_Float16)ftanh(c[v] + bv);
            }
        });
        __syncthreads();
        dense<128,128>(actA, W + WO2, lane, rowtile, colgroup, [&](v8f c, int nt){
            int n = nt * 16 + nl; float bv = bz.p[2][n];
#pragma unroll
            for (int v = 0; v < 8; ++v) {
                int row = rowtile * 16 + hi * 8 + v;
                actB[row * AS + n] = (_Float16)ftanh(c[v] + bv);
            }
        });
        __syncthreads();
        // hidden_ode = h + mlp(h)*dt   (stored back into h buffer)
        dense<128,64>(actB, W + WO3, lane, rowtile, colgroup, [&](v8f c, int nt){
            int n = nt * 16 + nl; float bv = bz.p[3][n];
#pragma unroll
            for (int v = 0; v < 8; ++v) {
                int row = rowtile * 16 + hi * 8 + v;
                h[row * LAT + n] = h[row * LAT + n] + (c[v] + bv) * dt;
            }
        });
        __syncthreads();

        // ---- cat = [hidden_ode, hstd, obs] -> actA (f16, 160 cols) ----------
        for (int i = tid; i < MT * 160; i += 256) {
            int row = i / 160, col = i - row * 160;
            float v = (col < 64) ? h[row * LAT + col]
                    : (col < 128) ? hstd[row * LAT + col - 64]
                                  : obs[row * IN_ + col - 128];
            actA[row * AS + col] = (_Float16)v;
        }
        __syncthreads();

        // ---- update gate u = sigmoid(W1 tanh(W0 cat + b0) + b1) ------------
        dense<160,128>(actA, W + WO4, lane, rowtile, colgroup, [&](v8f c, int nt){
            int n = nt * 16 + nl; float bv = bz.p[4][n];
#pragma unroll
            for (int v = 0; v < 8; ++v) {
                int row = rowtile * 16 + hi * 8 + v;
                actB[row * AS + n] = (_Float16)ftanh(c[v] + bv);
            }
        });
        __syncthreads();
        dense<128,64>(actB, W + WO5, lane, rowtile, colgroup, [&](v8f c, int nt){
            int n = nt * 16 + nl; float bv = bz.p[5][n];
#pragma unroll
            for (int v = 0; v < 8; ++v) {
                int row = rowtile * 16 + hi * 8 + v;
                u16[row * LAT + n] = (_Float16)fsigmoid(c[v] + bv);
            }
        });
        __syncthreads();

        // ---- reset gate r ---------------------------------------------------
        dense<160,128>(actA, W + WO6, lane, rowtile, colgroup, [&](v8f c, int nt){
            int n = nt * 16 + nl; float bv = bz.p[6][n];
#pragma unroll
            for (int v = 0; v < 8; ++v) {
                int row = rowtile * 16 + hi * 8 + v;
                actB[row * AS + n] = (_Float16)ftanh(c[v] + bv);
            }
        });
        __syncthreads();
        dense<128,64>(actB, W + WO7, lane, rowtile, colgroup, [&](v8f c, int nt){
            int n = nt * 16 + nl; float bv = bz.p[7][n];
#pragma unroll
            for (int v = 0; v < 8; ++v) {
                int row = rowtile * 16 + hi * 8 + v;
                r16[row * LAT + n] = (_Float16)fsigmoid(c[v] + bv);
            }
        });
        __syncthreads();

        // ---- cat2 = [hidden_ode*r, hstd*r, obs] -> actB ---------------------
        for (int i = tid; i < MT * 160; i += 256) {
            int row = i / 160, col = i - row * 160;
            float v;
            if (col < 64)       v = h[row * LAT + col] * (float)r16[row * LAT + col];
            else if (col < 128) v = hstd[row * LAT + col - 64] * (float)r16[row * LAT + col - 64];
            else                v = obs[row * IN_ + col - 128];
            actB[row * AS + col] = (_Float16)v;
        }
        __syncthreads();

        // ---- new state/std --------------------------------------------------
        dense<160,128>(actB, W + WO8, lane, rowtile, colgroup, [&](v8f c, int nt){
            int n = nt * 16 + nl; float bv = bz.p[8][n];
#pragma unroll
            for (int v = 0; v < 8; ++v) {
                int row = rowtile * 16 + hi * 8 + v;
                actA[row * AS + n] = (_Float16)ftanh(c[v] + bv);
            }
        });
        __syncthreads();
        dense<128,128>(actA, W + WO9, lane, rowtile, colgroup, [&](v8f c, int nt){
            int n = nt * 16 + nl; float bv = bz.p[9][n];
#pragma unroll
            for (int v = 0; v < 8; ++v) {
                int row = rowtile * 16 + hi * 8 + v;
                float x = c[v] + bv;
                if (n < 64) ns16[row * LAT + n]        = (_Float16)x;
                else        nd16[row * LAT + (n - 64)] = (_Float16)fabsf(x);
            }
        });
        // per-row observation mask and mask_first (overlap with new2 epilogue)
        if (tid < MT) {
            const float* mp = mask + (size_t)(b0 + tid) * (T_ * IN_) + (size_t)(t + 1) * IN_;
            float s = 0.f;
#pragma unroll
            for (int j = 0; j < IN_; ++j) s += mp[j];
            mrow[tid] = (s > 0.f) ? 1.f : 0.f;
            mfro[tid] = mask_first[(size_t)(b0 + tid) * TSTEPS + t];
        }
        __syncthreads();

        // ---- GRU combine + masking; new_y (f16) -> actB for decoder --------
        for (int i = tid; i < MT * LAT; i += 256) {
            int row = i >> 6;
            float u  = (float)u16[i];
            float ho = h[i], hs = hstd[i];
            float ny  = (1.f - u) * (float)ns16[i] + u * ho;
            float nys = (1.f - u) * (float)nd16[i] + u * hs;
            float m  = mrow[row], mf = mfro[row];
            h[i]    = mf * (m * ny  + (1.f - m) * ho);
            hstd[i] = mf * (m * nys + (1.f - m) * hs);
            actB[row * AS + (i & 63)] = (_Float16)ny;   // out_h = new_y (pre-mask)
        }
        // prev_obs = prev_obs*(1-mask) + mask*data
        for (int i = tid; i < MT * IN_; i += 256) {
            int row = i >> 5, c = i & 31;
            size_t gi = (size_t)(b0 + row) * (T_ * IN_) + (size_t)(t + 1) * IN_ + c;
            float mk = mask[gi];
            obs[i] = obs[i] * (1.f - mk) + mk * data[gi];
        }
        __syncthreads();

        // ---- decoder: 64->128 tanh -> 64 (mean | softplus std) -------------
        dense<64,128>(actB, W + WO10, lane, rowtile, colgroup, [&](v8f c, int nt){
            int n = nt * 16 + nl; float bv = bz.p[10][n];
#pragma unroll
            for (int v = 0; v < 8; ++v) {
                int row = rowtile * 16 + hi * 8 + v;
                actA[row * AS + n] = (_Float16)ftanh(c[v] + bv);
            }
        });
        __syncthreads();
        dense<128,64>(actA, W + WO11, lane, rowtile, colgroup, [&](v8f c, int nt){
            int n = nt * 16 + nl; float bv = bz.p[11][n];
#pragma unroll
            for (int v = 0; v < 8; ++v) {
                int row = rowtile * 16 + hi * 8 + v;
                int bg  = b0 + row;
                float x = c[v] + bv;
                size_t oi = ((size_t)bg * TSTEPS + t) * IN_;
                if (n < 32) out[oi + n] = x;                         // mean
                else        out[OUTOFF + oi + (n - 32)] = fsoftplus(x);
            }
        });
        __syncthreads();   // actA/actB reused next step
    }
}

// ---------------------------------------------------------------------------
// Host launcher.  Detects param-leaf flattening order from in_sizes.
// ---------------------------------------------------------------------------
extern "C" void kernel_launch(void* const* d_in, const int* in_sizes, int n_in,
                              void* d_out, int out_size, void* d_ws, size_t ws_size,
                              hipStream_t stream)
{
    (void)out_size; (void)ws_size;
    // top-level: insertion (data,mask,mask_first,time_steps,params) or
    // alphabetical (data,mask,mask_first,params...,time_steps)
    int pbase, tsi;
    if (in_sizes[3] == 256) { pbase = 4; tsi = 3; }
    else                    { pbase = 3; tsi = n_in - 1; }
    // inner dict: insertion (ode,update,reset,new,dec) -> leaf[pbase+2]=ode W1 (16384)
    //             alphabetical (dec,new,ode,reset,update) -> leaf[pbase+2]=dec W1 (8192)
    int ob, ub, rb, nb, db;
    if (in_sizes[pbase + 2] == 16384) {
        ob = pbase; ub = pbase + 8; rb = pbase + 12; nb = pbase + 16; db = pbase + 20;
    } else {
        db = pbase; nb = pbase + 4; ob = pbase + 8; rb = pbase + 16; ub = pbase + 20;
    }

    const float* data = (const float*)d_in[0];
    const float* mask = (const float*)d_in[1];
    const float* mfir = (const float*)d_in[2];
    const float* ts   = (const float*)d_in[tsi];
    _Float16* W = (_Float16*)d_ws;

    struct LW { int src; int off; int K; int N; };
    const LW lw[12] = {
        {ob + 0, WO0,  64, 128}, {ob + 2, WO1, 128, 128},
        {ob + 4, WO2, 128, 128}, {ob + 6, WO3, 128,  64},
        {ub + 0, WO4, 160, 128}, {ub + 2, WO5, 128,  64},
        {rb + 0, WO6, 160, 128}, {rb + 2, WO7, 128,  64},
        {nb + 0, WO8, 160, 128}, {nb + 2, WO9, 128, 128},
        {db + 0, WO10, 64, 128}, {db + 2, WO11, 128, 64},
    };
    Biases bz;
    const int bidx[12] = {ob + 1, ob + 3, ob + 5, ob + 7,
                          ub + 1, ub + 3, rb + 1, rb + 3,
                          nb + 1, nb + 3, db + 1, db + 3};
    for (int i = 0; i < 12; ++i) bz.p[i] = (const float*)d_in[bidx[i]];

    for (int i = 0; i < 12; ++i) {
        int total = lw[i].K * lw[i].N;
        prep_weights<<<(total + 255) / 256, 256, 0, stream>>>(
            (const float*)d_in[lw[i].src], W + lw[i].off, lw[i].K, lw[i].N);
    }
    ode_rnn_main<<<NBLK, 256, 0, stream>>>(data, mask, mfir, ts, bz, W, (float*)d_out);
}
